// SurpriseBasedMemory_32487132626975
// MI455X (gfx1250) — matrix operand order repaired and checked by reading, not compiled, not run on hardware
//
#include <hip/hip_runtime.h>
#include <hip/hip_bf16.h>

// ---------------------------------------------------------------------------
// SurpriseBasedMemory for MI455X (gfx1250, wave32, WMMA bf16 path)
// B=4 S=8192 H=512 M=1024 K=64  -> 32768 tokens
// ---------------------------------------------------------------------------

typedef __attribute__((ext_vector_type(16))) __bf16 v16bf;
typedef __attribute__((ext_vector_type(8)))  float  v8f;

union FragBF16 { uint4 u[2]; v16bf v; };

__device__ __forceinline__ unsigned short f32_to_bf16(float f) {
    unsigned int u = __float_as_uint(f);
    unsigned int r = u + 0x7FFFu + ((u >> 16) & 1u);   // round-to-nearest-even
    return (unsigned short)(r >> 16);
}

// ---------------------------------------------------------------------------
// Kernel 1: elementwise fp32 -> bf16 (weights / memory bank)
// ---------------------------------------------------------------------------
__global__ __launch_bounds__(256) void cvt_bf16_kernel(const float* __restrict__ src,
                                                       unsigned short* __restrict__ dst,
                                                       int n) {
    int i = blockIdx.x * 256 + threadIdx.x;
    if (i < n) dst[i] = f32_to_bf16(src[i]);
}

// ---------------------------------------------------------------------------
// Kernel 2: fused LayerNorm + surprise + bf16 cast.  One token per wave32,
// 16 elements per lane (H=512).
// ---------------------------------------------------------------------------
__global__ __launch_bounds__(256) void ln_surprise_kernel(const float* __restrict__ hidden,
                                                          const float* __restrict__ gamma,
                                                          const float* __restrict__ beta,
                                                          unsigned short* __restrict__ norm_bf,
                                                          float* __restrict__ surprise) {
    const int lane  = threadIdx.x & 31;
    const int wave  = threadIdx.x >> 5;
    const int token = blockIdx.x * 8 + wave;

    const float* x = hidden + (size_t)token * 512 + lane * 16;
    float v[16];
    const float4* xp = (const float4*)x;
#pragma unroll
    for (int q = 0; q < 4; ++q) {
        float4 t = xp[q];
        v[4*q+0] = t.x; v[4*q+1] = t.y; v[4*q+2] = t.z; v[4*q+3] = t.w;
    }

    float s = 0.f;
#pragma unroll
    for (int i = 0; i < 16; ++i) s += v[i];
#pragma unroll
    for (int off = 16; off > 0; off >>= 1) s += __shfl_xor(s, off, 32);
    const float mu = s * (1.0f / 512.0f);

    float vs = 0.f;
#pragma unroll
    for (int i = 0; i < 16; ++i) { float d = v[i] - mu; vs += d * d; }
#pragma unroll
    for (int off = 16; off > 0; off >>= 1) vs += __shfl_xor(vs, off, 32);
    const float rstd = rsqrtf(vs * (1.0f / 512.0f) + 1e-12f);

    const float4* gp = (const float4*)(gamma + lane * 16);
    const float4* bp = (const float4*)(beta  + lane * 16);
    float g[16], b[16];
#pragma unroll
    for (int q = 0; q < 4; ++q) {
        float4 tg = gp[q], tb = bp[q];
        g[4*q+0] = tg.x; g[4*q+1] = tg.y; g[4*q+2] = tg.z; g[4*q+3] = tg.w;
        b[4*q+0] = tb.x; b[4*q+1] = tb.y; b[4*q+2] = tb.z; b[4*q+3] = tb.w;
    }

    union { unsigned short s16[16]; uint4 u[2]; } ob;
    float asum = 0.f;
#pragma unroll
    for (int i = 0; i < 16; ++i) {
        float n = (v[i] - mu) * rstd * g[i] + b[i];
        asum += fabsf(n);
        ob.s16[i] = f32_to_bf16(n);
    }
#pragma unroll
    for (int off = 16; off > 0; off >>= 1) asum += __shfl_xor(asum, off, 32);
    if (lane == 0) surprise[token] = asum * (1.0f / 512.0f);

    uint4* op = (uint4*)(norm_bf + (size_t)token * 512 + lane * 16);
    op[0] = ob.u[0];
    op[1] = ob.u[1];
}

// ---------------------------------------------------------------------------
// Kernel 3: single-block selection.
//  - top-64 surprise tokens (descending, tie -> lower index), valid = > 0.8
//  - bottom-64 importance slots (ascending, tie -> lower index)
// Mutates surprise[] scratch (regenerated every launch by ln_surprise).
// ---------------------------------------------------------------------------
__global__ __launch_bounds__(1024) void select_kernel(float* __restrict__ surprise,
                                                      const float* __restrict__ importance,
                                                      int* __restrict__ top_idx,
                                                      int* __restrict__ slot_idx,
                                                      int* __restrict__ valid) {
    __shared__ float sval[1024];
    __shared__ int   sidx[1024];
    __shared__ float simp[1024];
    const int tid = threadIdx.x;

    // ---- phase 1: top-64 of surprise[32768] ----
    for (int i = 0; i < 64; ++i) {
        float bv = -3.4e38f; int bi = 0x7fffffff;
        for (int j = tid; j < 32768; j += 1024) {
            float v = surprise[j];
            if (v > bv || (v == bv && j < bi)) { bv = v; bi = j; }
        }
        sval[tid] = bv; sidx[tid] = bi; __syncthreads();
        for (int st = 512; st > 0; st >>= 1) {
            if (tid < st) {
                float ov = sval[tid + st]; int oi = sidx[tid + st];
                if (ov > sval[tid] || (ov == sval[tid] && oi < sidx[tid])) {
                    sval[tid] = ov; sidx[tid] = oi;
                }
            }
            __syncthreads();
        }
        if (tid == 0) {
            top_idx[i] = sidx[0];
            valid[i]   = (sval[0] > 0.8f) ? 1 : 0;
            surprise[sidx[0]] = -3.4e38f;
        }
        __syncthreads();
    }

    // ---- phase 2: bottom-64 of importance[1024] ----
    simp[tid] = importance[tid];
    __syncthreads();
    for (int i = 0; i < 64; ++i) {
        sval[tid] = simp[tid]; sidx[tid] = tid; __syncthreads();
        for (int st = 512; st > 0; st >>= 1) {
            if (tid < st) {
                float ov = sval[tid + st]; int oi = sidx[tid + st];
                if (ov < sval[tid] || (ov == sval[tid] && oi < sidx[tid])) {
                    sval[tid] = ov; sidx[tid] = oi;
                }
            }
            __syncthreads();
        }
        if (tid == 0) { slot_idx[i] = sidx[0]; simp[sidx[0]] = 3.4e38f; }
        __syncthreads();
    }
}

// ---------------------------------------------------------------------------
// Kernel 4: scatter selected norm rows into bf16 memory bank.
// grid = 64 blocks x 128 threads, 8 B/thread -> 1024 B row.
// ---------------------------------------------------------------------------
__global__ __launch_bounds__(128) void scatter_kernel(const int* __restrict__ top_idx,
                                                      const int* __restrict__ slot_idx,
                                                      const int* __restrict__ valid,
                                                      const unsigned short* __restrict__ norm_bf,
                                                      unsigned short* __restrict__ mem_bf) {
    const int i = blockIdx.x;
    if (!valid[i]) return;
    const int t = top_idx[i], s = slot_idx[i];
    const uint2* src = (const uint2*)(norm_bf + (size_t)t * 512);
    uint2*       dst = (uint2*)(mem_bf + (size_t)s * 512);
    dst[threadIdx.x] = src[threadIdx.x];
}

// ---------------------------------------------------------------------------
// Kernel 5: double-buffered LDS-staged bf16 WMMA GEMM.
//   C[M,N] = A[M,K] * B + bias (+res)
//   B_NMAJOR=1 : B stored as Bn[n][k] (ldb=K)  (Wq, Wo, mem used as B^T)
//   B_NMAJOR=0 : B stored as Bk[k][n] (ldb=N)  (mem for attn@mem)
// Tile 128x64, BK=32; 8 waves (4 along M x 2 along N); each wave: 32x32
//   -> four v_wmma_f32_16x16x32_bf16 per K-step, ONE barrier per K-step,
//   next tile's global loads issued before the WMMAs (overlap).
// M%128==0, N%64==0, K%32==0 (holds for all call sites).
// ---------------------------------------------------------------------------
#define LDSP 40   // padded row pitch (bf16 elems) for both LDS tiles

template<int B_NMAJOR, int OUT_BF16, int HAS_BIAS, int HAS_RES>
__global__ __launch_bounds__(256) void gemm_bf16_kernel(const unsigned short* __restrict__ A, int lda,
                                                        const unsigned short* __restrict__ B, int ldb,
                                                        const float* __restrict__ bias,
                                                        const float* __restrict__ residual,
                                                        void* __restrict__ out,
                                                        int M, int N, int K) {
    __shared__ __align__(16) unsigned short As[2][128 * LDSP];
    __shared__ __align__(16) unsigned short Bs[2][64 * LDSP];   // N-major: Bs[n][k]

    const int tid  = threadIdx.x;
    const int lane = tid & 31;
    const int wave = tid >> 5;
    const int wm = wave & 3;     // 4 waves along M (32 rows each)
    const int wn = wave >> 2;    // 2 waves along N (32 cols each)
    const int m0 = blockIdx.y * 128;
    const int n0 = blockIdx.x * 64;

    // ---- staging source pointers (advance by k0 elements each step) ----
    const int arow = tid >> 2;            // 0..63
    const int achk = tid & 3;             // 0..3 (k chunks of 8)
    const unsigned short* pa0 = A + (size_t)(m0 + arow) * lda + achk * 8;
    const unsigned short* pa1 = A + (size_t)(m0 + arow + 64) * lda + achk * 8;
    const unsigned short* pb;
    int brow, bchk;
    if (B_NMAJOR) {
        brow = tid >> 2; bchk = tid & 3;                         // 64 n-rows x 4 chunks
        pb = B + (size_t)(n0 + brow) * ldb + bchk * 8;
    } else {
        brow = tid >> 3; bchk = tid & 7;                         // 32 k-rows x 8 chunks
        pb = B + (size_t)brow * ldb + n0 + bchk * 8;
    }

    uint4 ra0, ra1, rb;
    auto load_tiles = [&](int k0) {
        ra0 = *(const uint4*)(pa0 + k0);
        ra1 = *(const uint4*)(pa1 + k0);
        if (B_NMAJOR) rb = *(const uint4*)(pb + k0);
        else          rb = *(const uint4*)(pb + (size_t)k0 * ldb);
    };
    auto store_tiles = [&](int buf) {
        *(uint4*)(&As[buf][arow * LDSP + achk * 8])        = ra0;
        *(uint4*)(&As[buf][(arow + 64) * LDSP + achk * 8]) = ra1;
        if (B_NMAJOR) {
            *(uint4*)(&Bs[buf][brow * LDSP + bchk * 8]) = rb;
        } else {
            const unsigned short* pv = (const unsigned short*)&rb;
#pragma unroll
            for (int j = 0; j < 8; ++j)
                Bs[buf][(bchk * 8 + j) * LDSP + brow] = pv[j];   // transpose on store
        }
    };

    v8f acc00 = {}, acc01 = {}, acc10 = {}, acc11 = {};

    // prologue: fill buffer 0
    load_tiles(0);
    store_tiles(0);
    __syncthreads();

    const int hk  = lane >> 4;     // lane half
    const int l16 = lane & 15;
    const int mlo0 = (wm * 32 + l16) * LDSP;
    const int mlo1 = (wm * 32 + 16 + l16) * LDSP;
    const int nlo0 = (wn * 32 + l16) * LDSP;
    const int nlo1 = (wn * 32 + 16 + l16) * LDSP;

    int it = 0;
    for (int k0 = 0; k0 < K; k0 += 32, ++it) {
        const int  cur  = it & 1;
        const bool more = (k0 + 32) < K;

        // issue next tile's global loads NOW (waited on only at store_tiles,
        // after the WMMAs -> load latency overlaps matrix math)
        if (more) load_tiles(k0 + 32);
        if (k0 + 64 < K) {
            __builtin_prefetch(pa0 + k0 + 64, 0, 1);
            __builtin_prefetch(pa1 + k0 + 64, 0, 1);
            if (B_NMAJOR) __builtin_prefetch(pb + k0 + 64, 0, 1);
            else          __builtin_prefetch(pb + (size_t)(k0 + 64) * ldb, 0, 1);
        }

        // ---- fragments per ISA 7.12.2 (wave32, 16-bit A 16x32 / B 32x16) ----
        const unsigned short* asb = &As[cur][0];
        const unsigned short* bsb = &Bs[cur][0];
        FragBF16 fa0, fa1, fb0, fb1;
        fa0.u[0] = *(const uint4*)(&asb[mlo0 + hk * 8]);        // e0..7 : K = hk*8 + e
        fa0.u[1] = *(const uint4*)(&asb[mlo0 + 16 + hk * 8]);   // e8..15: K = 16 + hk*8 + e
        fa1.u[0] = *(const uint4*)(&asb[mlo1 + hk * 8]);
        fa1.u[1] = *(const uint4*)(&asb[mlo1 + 16 + hk * 8]);
        fb0.u[0] = *(const uint4*)(&bsb[nlo0 + hk * 8]);        // K = hk*8 + e, 16 + hk*8 + e
        fb0.u[1] = *(const uint4*)(&bsb[nlo0 + 16 + hk * 8]);
        fb1.u[0] = *(const uint4*)(&bsb[nlo1 + hk * 8]);
        fb1.u[1] = *(const uint4*)(&bsb[nlo1 + 16 + hk * 8]);

        acc00 = __builtin_amdgcn_wmma_f32_16x16x32_bf16(false, fa0.v, false, fb0.v,
                                                        (short)0, acc00, false, false);
        acc01 = __builtin_amdgcn_wmma_f32_16x16x32_bf16(false, fa0.v, false, fb1.v,
                                                        (short)0, acc01, false, false);
        acc10 = __builtin_amdgcn_wmma_f32_16x16x32_bf16(false, fa1.v, false, fb0.v,
                                                        (short)0, acc10, false, false);
        acc11 = __builtin_amdgcn_wmma_f32_16x16x32_bf16(false, fa1.v, false, fb1.v,
                                                        (short)0, acc11, false, false);

        if (more) store_tiles(cur ^ 1);
        __syncthreads();            // single barrier per K-step
    }

    // ---- epilogue: C/D layout (lane half -> rows r / r+8, col = lane%16) ----
    auto emit = [&](int row, int col, float v) {
        if (HAS_BIAS) v += bias[col];
        if (HAS_RES)  v += residual[(size_t)row * N + col];
        if (OUT_BF16) ((unsigned short*)out)[(size_t)row * N + col] = f32_to_bf16(v);
        else          ((float*)out)[(size_t)row * N + col] = v;
    };
#pragma unroll
    for (int r = 0; r < 8; ++r) {
        const int row0 = m0 + wm * 32 + hk * 8 + r;   // rows of fa0 accums
        const int row1 = row0 + 16;                   // rows of fa1 accums
        const int c0   = n0 + wn * 32 + l16;
        const int c1   = c0 + 16;
        emit(row0, c0, acc00[r]);
        emit(row0, c1, acc01[r]);
        emit(row1, c0, acc10[r]);
        emit(row1, c1, acc11[r]);
    }
}

// ---------------------------------------------------------------------------
// Kernel 6: row softmax (fp32 scores -> bf16 probs), ncol = 1024.
// ---------------------------------------------------------------------------
__global__ __launch_bounds__(256) void softmax_rows_kernel(const float* __restrict__ scores,
                                                           unsigned short* __restrict__ probs,
                                                           int ncol) {
    __shared__ float red[256];
    const int row = blockIdx.x;
    const float* s = scores + (size_t)row * ncol;

    float lmax = -3.4e38f;
    for (int j = threadIdx.x; j < ncol; j += 256) lmax = fmaxf(lmax, s[j]);
    red[threadIdx.x] = lmax; __syncthreads();
    for (int st = 128; st > 0; st >>= 1) {
        if (threadIdx.x < st) red[threadIdx.x] = fmaxf(red[threadIdx.x], red[threadIdx.x + st]);
        __syncthreads();
    }
    const float m = red[0];
    __syncthreads();

    float lsum = 0.f;
    for (int j = threadIdx.x; j < ncol; j += 256) lsum += __expf(s[j] - m);
    red[threadIdx.x] = lsum; __syncthreads();
    for (int st = 128; st > 0; st >>= 1) {
        if (threadIdx.x < st) red[threadIdx.x] += red[threadIdx.x + st];
        __syncthreads();
    }
    const float inv = 1.0f / red[0];

    unsigned short* p = probs + (size_t)row * ncol;
    for (int j = threadIdx.x; j < ncol; j += 256)
        p[j] = f32_to_bf16(__expf(s[j] - m) * inv);
}

// ---------------------------------------------------------------------------
// Host side
// ---------------------------------------------------------------------------
extern "C" void kernel_launch(void* const* d_in, const int* in_sizes, int n_in,
                              void* d_out, int out_size, void* d_ws, size_t ws_size,
                              hipStream_t stream) {
    (void)in_sizes; (void)n_in; (void)out_size; (void)ws_size;

    const float* hidden     = (const float*)d_in[0];   // [4,8192,512]
    const float* memory     = (const float*)d_in[1];   // [1,1024,512]
    const float* importance = (const float*)d_in[2];   // [1,1024]
    const float* gamma      = (const float*)d_in[3];   // [512]
    const float* beta       = (const float*)d_in[4];   // [512]
    const float* Wq         = (const float*)d_in[5];   // [512,512]
    const float* bq         = (const float*)d_in[6];   // [512]
    const float* Wo         = (const float*)d_in[7];   // [512,512]
    const float* bo         = (const float*)d_in[8];   // [512]
    float* out = (float*)d_out;                        // [4,8192,512] fp32

    constexpr int NT = 32768;   // tokens
    constexpr int H  = 512;
    constexpr int M  = 1024;
    constexpr int CH = 8192;    // token chunk for attention
    constexpr int NCHUNK = NT / CH;

    // workspace carve-out (~150 MB total)
    char* ws = (char*)d_ws;
    size_t cur = 0;
    auto alloc = [&](size_t bytes) -> void* {
        void* p = ws + cur;
        cur = (cur + bytes + 255) & ~(size_t)255;
        return p;
    };
    unsigned short* norm_bf  = (unsigned short*)alloc((size_t)NT * H * 2);   // 32 MB
    unsigned short* q_bf     = (unsigned short*)alloc((size_t)NT * H * 2);   // 32 MB
    unsigned short* ret_bf   = (unsigned short*)alloc((size_t)NT * H * 2);   // 32 MB
    unsigned short* mem_bf   = (unsigned short*)alloc((size_t)M  * H * 2);   //  1 MB
    unsigned short* wq_bf    = (unsigned short*)alloc((size_t)H  * H * 2);   // .5 MB
    unsigned short* wo_bf    = (unsigned short*)alloc((size_t)H  * H * 2);   // .5 MB
    float*          surprise = (float*)alloc((size_t)NT * 4);                // 128 KB
    int*            top_idx  = (int*)alloc(64 * 4);
    int*            slot_idx = (int*)alloc(64 * 4);
    int*            valid    = (int*)alloc(64 * 4);
    float*          scores   = (float*)alloc((size_t)CH * M * 4);            // 32 MB
    unsigned short* probs    = (unsigned short*)alloc((size_t)CH * M * 2);   // 16 MB

    // 1) precision conversion of static operands
    cvt_bf16_kernel<<<(H * H + 255) / 256, 256, 0, stream>>>(Wq, wq_bf, H * H);
    cvt_bf16_kernel<<<(H * H + 255) / 256, 256, 0, stream>>>(Wo, wo_bf, H * H);
    cvt_bf16_kernel<<<(M * H + 255) / 256, 256, 0, stream>>>(memory, mem_bf, M * H);

    // 2) LayerNorm + surprise + bf16 norm
    ln_surprise_kernel<<<NT / 8, 256, 0, stream>>>(hidden, gamma, beta, norm_bf, surprise);

    // 3) top-64 surprise tokens / bottom-64 importance slots
    select_kernel<<<1, 1024, 0, stream>>>(surprise, importance, top_idx, slot_idx, valid);

    // 4) scatter valid rows into memory bank
    scatter_kernel<<<64, 128, 0, stream>>>(top_idx, slot_idx, valid, norm_bf, mem_bf);

    // 5) query projection: Q = norm @ Wq^T + bq  (bf16 out)
    gemm_bf16_kernel<1, 1, 1, 0><<<dim3(H / 64, NT / 128), 256, 0, stream>>>(
        norm_bf, H, wq_bf, H, bq, nullptr, q_bf, NT, H, H);

    // 6) attention in token chunks: scores -> softmax -> retrieved
    for (int c = 0; c < NCHUNK; ++c) {
        const unsigned short* qc = q_bf + (size_t)c * CH * H;
        unsigned short* rc = ret_bf + (size_t)c * CH * H;

        // scores[CH,M] = Qc @ mem^T   (mem rows are N-major for this GEMM)
        gemm_bf16_kernel<1, 0, 0, 0><<<dim3(M / 64, CH / 128), 256, 0, stream>>>(
            qc, H, mem_bf, H, nullptr, nullptr, scores, CH, M, H);

        softmax_rows_kernel<<<CH, 256, 0, stream>>>(scores, probs, M);

        // retrieved[CH,H] = probs @ mem   (mem is K-major here)
        gemm_bf16_kernel<0, 1, 0, 0><<<dim3(H / 64, CH / 128), 256, 0, stream>>>(
            probs, M, mem_bf, H, nullptr, nullptr, rc, CH, H, M);
    }

    // 7) output projection + bias + residual (fp32 out)
    gemm_bf16_kernel<1, 0, 1, 1><<<dim3(H / 64, NT / 128), 256, 0, stream>>>(
        ret_bf, H, wo_bf, H, bo, hidden, out, NT, H, H);
}
